// PAM_Module_42477226558032
// MI455X (gfx1250) — compile-verified, matrix operand order checked by reading, not run on hardware
//
#include <hip/hip_runtime.h>

typedef __attribute__((ext_vector_type(16))) _Float16 v16h;
typedef __attribute__((ext_vector_type(8)))  _Float16 v8h;
typedef __attribute__((ext_vector_type(16))) __bf16   v16bf;
typedef __attribute__((ext_vector_type(8)))  __bf16   v8bf;
typedef __attribute__((ext_vector_type(8)))  float    v8f;

#define BATCH 8
#define CIN   64
#define HIN   128
#define WIN   128
#define NPIX  4096   // 64*64 pooled pixels
#define DQK   32

// Load 16 f16 in the WMMA 16-bit operand per-lane pattern:
// vec[0..7] = base[kb..kb+7], vec[8..15] = base[kb+16..kb+23]
__device__ __forceinline__ v16h load_op16h(const _Float16* base, int kb) {
  v8h lo = *(const v8h*)(base + kb);
  v8h hi = *(const v8h*)(base + kb + 16);
  v16h r;
  #pragma unroll
  for (int i = 0; i < 8; ++i) { r[i] = lo[i]; r[i + 8] = hi[i]; }
  return r;
}
__device__ __forceinline__ v16bf load_op16bf(const __bf16* base, int kb) {
  v8bf lo = *(const v8bf*)(base + kb);
  v8bf hi = *(const v8bf*)(base + kb + 16);
  v16bf r;
  #pragma unroll
  for (int i = 0; i < 8; ++i) { r[i] = lo[i]; r[i + 8] = hi[i]; }
  return r;
}

// ---------------- Kernel 0a: x (NCHW f32) -> xh (NHWC bf16) -----------------
__global__ __launch_bounds__(256) void nhwc_kernel(
    const float* __restrict__ x, __bf16* __restrict__ xh)
{
  int idx = blockIdx.x * 256 + threadIdx.x;          // [b][h][w][c]
  if (idx >= BATCH * HIN * WIN * CIN) return;
  int c = idx & 63, w = (idx >> 6) & 127, h = (idx >> 13) & 127, b = idx >> 20;
  xh[idx] = (__bf16)x[(((size_t)b * CIN + c) * HIN + h) * WIN + w];
}

// ---------------- Kernel 0b: pool_w (OIHW) -> wT2 [co][tap*64+ci] bf16 ------
__global__ __launch_bounds__(256) void wtrans_kernel(
    const float* __restrict__ pw, __bf16* __restrict__ wT2)
{
  int idx = blockIdx.x * 256 + threadIdx.x;          // co*576 + K
  if (idx >= CIN * 576) return;
  int co = idx / 576, K = idx - co * 576;
  int tap = K >> 6, ci = K & 63;
  wT2[idx] = (__bf16)pw[((size_t)co * CIN + ci) * 9 + tap];
}

// ---------------- Kernel 1: 3x3 stride-2 conv as implicit-GEMM WMMA ---------
__global__ __launch_bounds__(128) void conv_wmma_kernel(
    const __bf16* __restrict__ xh, const __bf16* __restrict__ wT2,
    const float* __restrict__ pb, float* __restrict__ xp)
{
  const int lane = threadIdx.x & 31, warp = threadIdx.x >> 5;
  const int halfW = lane >> 4, l16 = lane & 15, kb = halfW * 8;
  const int pbase = (blockIdx.x * 4 + warp) * 16;    // global pixel id base
  const int b = pbase >> 12;
  const int nbase = pbase & 4095;
  const int n = nbase + l16;                         // lane's A-row pixel
  const int oh = n >> 6, ow = n & 63;

  v8f Oc[4];
  const v8f zc = {};
  #pragma unroll
  for (int cc = 0; cc < 4; ++cc) Oc[cc] = zc;

  #pragma unroll
  for (int kc = 0; kc < 18; ++kc) {
    const int tap = kc >> 1, c0 = (kc & 1) * 32;
    const int kh = tap / 3, kw = tap % 3;
    const int ih = oh * 2 + kh - 1, iw = ow * 2 + kw - 1;
    v16bf aX;
    if (ih >= 0 && ih < HIN && iw >= 0 && iw < WIN) {
      const __bf16* row = xh + ((size_t)((b * HIN + ih) * WIN + iw)) * CIN + c0;
      aX = load_op16bf(row, kb);
    } else {
      v16bf z = {};
      aX = z;
    }
    #pragma unroll
    for (int cc = 0; cc < 4; ++cc) {
      const __bf16* wrow = wT2 + (size_t)(cc * 16 + l16) * 576 + tap * 64 + c0;
      v16bf bW = load_op16bf(wrow, kb);
      Oc[cc] = __builtin_amdgcn_wmma_f32_16x16x32_bf16(false, aX, false, bW,
                                                       (short)0, Oc[cc],
                                                       false, false);
    }
  }
  #pragma unroll
  for (int cc = 0; cc < 4; ++cc) {
    const int co = cc * 16 + l16;
    const float bias = pb[co];
    #pragma unroll
    for (int r = 0; r < 8; ++r) {
      const int np = nbase + r + 8 * halfW;
      xp[(size_t)(b * CIN + co) * NPIX + np] = Oc[cc][r] + bias;
    }
  }
}

// ---------------- Kernel 2: fused Q/K/V 1x1 projections -> f16 --------------
__global__ __launch_bounds__(128) void qkv_kernel(
    const float* __restrict__ xp,
    const float* __restrict__ qw, const float* __restrict__ qb,
    const float* __restrict__ kw, const float* __restrict__ kb_,
    const float* __restrict__ vw, const float* __restrict__ vb,
    _Float16* __restrict__ qT, _Float16* __restrict__ kT,
    _Float16* __restrict__ vF)
{
  __shared__ float xcol[CIN];
  int t = threadIdx.x;
  int b = blockIdx.x >> 12;
  int n = blockIdx.x & (NPIX - 1);
  if (t < CIN) xcol[t] = xp[(size_t)(b * CIN + t) * NPIX + n];
  __syncthreads();
  const float* wrow; float bias;
  if (t < 32)      { wrow = qw + t * CIN;        bias = qb[t]; }
  else if (t < 64) { wrow = kw + (t - 32) * CIN; bias = kb_[t - 32]; }
  else             { wrow = vw + (t - 64) * CIN; bias = vb[t - 64]; }
  float acc = bias;
  #pragma unroll 8
  for (int c = 0; c < CIN; ++c) acc = fmaf(wrow[c], xcol[c], acc);
  if (t < 32)      qT[(size_t)(b * NPIX + n) * DQK + t]        = (_Float16)acc;
  else if (t < 64) kT[(size_t)(b * NPIX + n) * DQK + (t - 32)] = (_Float16)acc;
  else             vF[(size_t)(b * CIN + (t - 64)) * NPIX + n] = (_Float16)acc;
}

// ---------------- Kernel 3: flash attention, transposed-S formulation -------
// One wave: 32 queries. S^T = K Q^T so softmax is in-lane + one xor-16, and
// P lands directly in the A-operand layout for P*V^T. The V tile (shared by
// all 4 waves of the block; b is block-uniform) is staged into double-buffered
// LDS with GLOBAL_LOAD_ASYNC_TO_LDS_B128 (ASYNCcnt), overlapping the copy of
// tile mc+32 with compute on tile mc.
__global__ __launch_bounds__(128) void pam_attn_kernel(
    const _Float16* __restrict__ qT, const _Float16* __restrict__ kT,
    const _Float16* __restrict__ vF, float* __restrict__ attO)
{
  __shared__ _Float16 vtile[2][CIN * 32];      // 2 x 4KB V tiles
  const int tid  = threadIdx.x;
  const int lane = tid & 31;
  const int warp = tid >> 5;
  const int tile = blockIdx.x * 4 + warp;      // 0..1023
  const int b    = tile >> 7;                  // uniform across the block
  const int n0   = (tile & 127) << 5;
  const int halfW = lane >> 4;
  const int l16   = lane & 15;
  const int kb    = halfW * 8;

  const _Float16* kBase = kT + (size_t)b * NPIX * DQK;
  const _Float16* vBase = vF + (size_t)b * CIN * NPIX;

  // Cooperative async stage of V[:, mc..mc+31] -> vtile[buf] ([c][m] halves).
  // 256 x 16B lane-transfers spread over 128 threads x 2.
  auto stageV = [&](int mc, int buf) {
    #pragma unroll
    for (int i = 0; i < 2; ++i) {
      const int T = i * 128 + tid;             // 0..255
      const int c = T >> 2, mseg = T & 3;
      unsigned ldsOff = (unsigned)(unsigned long long)
          (&vtile[buf][c * 32 + mseg * 8]);
      unsigned long long gaddr = (unsigned long long)
          (vBase + (size_t)c * NPIX + mc + mseg * 8);
      asm volatile("global_load_async_to_lds_b128 %0, %1, off"
                   :: "v"(ldsOff), "v"(gaddr) : "memory");
    }
  };

  // Q as B-operands (column n = l16)
  v16h bQ[2];
  #pragma unroll
  for (int s = 0; s < 2; ++s)
    bQ[s] = load_op16h(qT + (size_t)(b * NPIX + n0 + s * 16 + l16) * DQK, kb);

  const v8f zc = {};
  v8f Oacc[2][4];
  #pragma unroll
  for (int s = 0; s < 2; ++s)
    #pragma unroll
    for (int c = 0; c < 4; ++c) Oacc[s][c] = zc;
  float mrun[2] = {-1e30f, -1e30f};
  float lrun[2] = {0.0f, 0.0f};

  stageV(0, 0);                                // first V tile in flight

  // Preload K A-operands for mc = 0
  v16h aK0 = load_op16h(kBase + (size_t)l16 * DQK, kb);
  v16h aK1 = load_op16h(kBase + (size_t)(16 + l16) * DQK, kb);

  asm volatile("s_wait_asynccnt 0x0" ::: "memory");
  __syncthreads();

  for (int mc = 0; mc < NPIX; mc += 32) {
    const int cur = (mc >> 5) & 1;
    if (mc + 32 < NPIX) stageV(mc + 32, cur ^ 1);   // overlap next V copy

    // S^T tiles for both sub-tiles, then immediately start next K loads
    v8f S0a = __builtin_amdgcn_wmma_f32_16x16x32_f16(false, aK0, false, bQ[0],
                                                     (short)0, zc, false, false);
    v8f S1a = __builtin_amdgcn_wmma_f32_16x16x32_f16(false, aK1, false, bQ[0],
                                                     (short)0, zc, false, false);
    v8f S0b = __builtin_amdgcn_wmma_f32_16x16x32_f16(false, aK0, false, bQ[1],
                                                     (short)0, zc, false, false);
    v8f S1b = __builtin_amdgcn_wmma_f32_16x16x32_f16(false, aK1, false, bQ[1],
                                                     (short)0, zc, false, false);
    if (mc + 32 < NPIX) {   // software pipeline: overwrite aK after last use
      aK0 = load_op16h(kBase + (size_t)(mc + 32 + l16) * DQK, kb);
      aK1 = load_op16h(kBase + (size_t)(mc + 48 + l16) * DQK, kb);
      __builtin_prefetch(kBase + (size_t)(mc + 64 + l16) * DQK, 0, 1);
    }

    // V B-operands from the staged LDS tile
    v16h bV[4];
    #pragma unroll
    for (int c = 0; c < 4; ++c)
      bV[c] = load_op16h(&vtile[cur][(c * 16 + l16) * 32], kb);

    #pragma unroll
    for (int s = 0; s < 2; ++s) {
      v8f& S0 = s ? S0b : S0a;
      v8f& S1 = s ? S1b : S1a;
      float t = fmaxf(S0[0], S1[0]);
      #pragma unroll
      for (int r = 1; r < 8; ++r) t = fmaxf(t, fmaxf(S0[r], S1[r]));
      t = fmaxf(t, __shfl_xor(t, 16));
      const float mn = fmaxf(mrun[s], t);
      const float alpha = __expf(mrun[s] - mn);
      mrun[s] = mn;
      float ps = 0.0f;
      v16h aP;
      #pragma unroll
      for (int r = 0; r < 8; ++r) {
        const float p0 = __expf(S0[r] - mn);
        const float p1 = __expf(S1[r] - mn);
        ps += p0 + p1;
        aP[r]     = (_Float16)p0;
        aP[r + 8] = (_Float16)p1;
      }
      ps += __shfl_xor(ps, 16);
      lrun[s] = alpha * lrun[s] + ps;
      float arow[8];
      #pragma unroll
      for (int r = 0; r < 8; ++r) arow[r] = __shfl(alpha, r + 8 * halfW, 16);
      #pragma unroll
      for (int c = 0; c < 4; ++c) {
        #pragma unroll
        for (int r = 0; r < 8; ++r) Oacc[s][c][r] *= arow[r];
        Oacc[s][c] = __builtin_amdgcn_wmma_f32_16x16x32_f16(
            false, aP, false, bV[c], (short)0, Oacc[s][c], false, false);
      }
    }

    // Own async copies done + all waves finished reading vtile[cur]
    asm volatile("s_wait_asynccnt 0x0" ::: "memory");
    __syncthreads();
  }

  #pragma unroll
  for (int s = 0; s < 2; ++s) {
    const float linv = 1.0f / lrun[s];
    #pragma unroll
    for (int r = 0; r < 8; ++r) {
      const float lr = __shfl(linv, r + 8 * halfW, 16);
      const int n = n0 + s * 16 + r + 8 * halfW;
      #pragma unroll
      for (int c = 0; c < 4; ++c)
        attO[(size_t)(b * CIN + c * 16 + l16) * NPIX + n] = Oacc[s][c][r] * lr;
    }
  }
}

// ---------------- Kernel 4: gamma * bilinear_up(att) + x --------------------
__global__ __launch_bounds__(256) void upsample_add_kernel(
    const float* __restrict__ att, const float* __restrict__ x,
    const float* __restrict__ gamma, float* __restrict__ out)
{
  int idx = blockIdx.x * 256 + threadIdx.x;
  if (idx >= BATCH * CIN * HIN * WIN) return;
  int wI = idx & 127, hI = (idx >> 7) & 127;
  int bc = idx >> 14;
  const float scale = 63.0f / 127.0f;    // align_corners=True, 64 -> 128
  float fh = hI * scale;
  int h0 = (int)fh; h0 = h0 > 62 ? 62 : h0;
  float th = fh - (float)h0;
  float fw = wI * scale;
  int w0 = (int)fw; w0 = w0 > 62 ? 62 : w0;
  float tw = fw - (float)w0;
  const float* a = att + (size_t)bc * NPIX;
  float a00 = a[h0 * 64 + w0],       a01 = a[h0 * 64 + w0 + 1];
  float a10 = a[(h0 + 1) * 64 + w0], a11 = a[(h0 + 1) * 64 + w0 + 1];
  float top = a00 + tw * (a01 - a00);
  float bot = a10 + tw * (a11 - a10);
  float val = top + th * (bot - top);
  out[idx] = gamma[0] * val + x[idx];
}

// ---------------------------------------------------------------------------
extern "C" void kernel_launch(void* const* d_in, const int* in_sizes, int n_in,
                              void* d_out, int out_size, void* d_ws, size_t ws_size,
                              hipStream_t stream) {
  const float* x   = (const float*)d_in[0];
  const float* pw  = (const float*)d_in[1];
  const float* pb  = (const float*)d_in[2];
  const float* qw  = (const float*)d_in[3];
  const float* qb  = (const float*)d_in[4];
  const float* kw  = (const float*)d_in[5];
  const float* kbv = (const float*)d_in[6];
  const float* vw  = (const float*)d_in[7];
  const float* vb  = (const float*)d_in[8];
  const float* gam = (const float*)d_in[9];
  float* out = (float*)d_out;

  // Workspace layout (~40 MB)
  float*    xp   = (float*)d_ws;                                    // 8 MB
  float*    attO = xp + (size_t)BATCH * CIN * NPIX;                 // 8 MB
  __bf16*   xh   = (__bf16*)(attO + (size_t)BATCH * CIN * NPIX);    // 16 MB
  __bf16*   wT2  = xh + (size_t)BATCH * HIN * WIN * CIN;            // 72 KB
  _Float16* qT   = (_Float16*)(wT2 + (size_t)CIN * 576);            // 2 MB
  _Float16* kT   = qT + (size_t)BATCH * NPIX * DQK;                 // 2 MB
  _Float16* vF   = kT + (size_t)BATCH * NPIX * DQK;                 // 4 MB

  nhwc_kernel<<<(BATCH * HIN * WIN * CIN) / 256, 256, 0, stream>>>(x, xh);
  wtrans_kernel<<<(CIN * 576 + 255) / 256, 256, 0, stream>>>(pw, wT2);
  conv_wmma_kernel<<<(BATCH * NPIX) / 64, 128, 0, stream>>>(xh, wT2, pb, xp);
  qkv_kernel<<<BATCH * NPIX, 128, 0, stream>>>(xp, qw, qb, kw, kbv, vw, vb,
                                               qT, kT, vF);
  pam_attn_kernel<<<(BATCH * (NPIX / 32)) / 4, 128, 0, stream>>>(qT, kT, vF, attO);
  upsample_add_kernel<<<(BATCH * CIN * HIN * WIN) / 256, 256, 0, stream>>>(
      attO, x, gam, out);
}
